// Linear_w_crossattention_59931973649013
// MI455X (gfx1250) — matrix operand order repaired and checked by reading, not compiled, not run on hardware
//
#include <hip/hip_runtime.h>

// ---------- types ----------
typedef __attribute__((ext_vector_type(16))) __bf16 v16bf;
typedef __attribute__((ext_vector_type(8)))  __bf16 v8bf;
typedef __attribute__((ext_vector_type(8)))  float  v8f;

#define NNODES 4096
#define DIM    128
#define HEADS  4
#define QKVN   1536          // 512 q | 512 k | 512 v
#define QSCALE 0.08838834764831845f   // 1/sqrt(128)

// workspace layout (bytes)
#define OFF_WT    ((size_t)0)                       // 1536*128 bf16  = 393216
#define OFF_BIAS  ((size_t)393216)                  // 1536 f32       = 6144
#define OFF_H     ((size_t)399360)                  // 4096*128 bf16  = 1 MiB
#define OFF_SKIP  ((size_t)1447936)                 // 4096*128 f32   = 2 MiB
#define OFF_Y     ((size_t)3545088)                 // 4096*1536 bf16 = 12 MiB
#define OFF_VT    ((size_t)16128000)                // 512*4096 bf16  = 4 MiB

static __device__ __forceinline__ v8f vzero8() {
    v8f z;
#pragma unroll
    for (int j = 0; j < 8; ++j) z[j] = 0.f;
    return z;
}

// ---------- K0: pack Wq|Wk|Wv transposed to bf16 col-major, fold q scale ----------
__global__ void pack_weights(const float* __restrict__ Wq, const float* __restrict__ Wk,
                             const float* __restrict__ Wv,
                             const float* __restrict__ bq, const float* __restrict__ bk,
                             const float* __restrict__ bv,
                             __bf16* __restrict__ WT, float* __restrict__ bias_all) {
    int idx = blockIdx.x * blockDim.x + threadIdx.x;
    if (idx >= QKVN * DIM) return;
    int c = idx >> 7;      // 0..1535 output column
    int k = idx & 127;     // 0..127  input channel
    float val;
    if (c < 512)       val = Wq[k * 512 + c] * QSCALE;
    else if (c < 1024) val = Wk[k * 512 + (c - 512)];
    else               val = Wv[k * 512 + (c - 1024)];
    WT[c * DIM + k] = (__bf16)val;
    if (k == 0) {
        float b;
        if (c < 512)       b = bq[c] * QSCALE;
        else if (c < 1024) b = bk[c - 512];
        else               b = bv[c - 1024];
        bias_all[c] = b;
    }
}

// ---------- K1: per row  h = LN(ELU(x@W1+b1)),  skip = h@Ws + bs ----------
__global__ __launch_bounds__(128) void row_ln_skip(
        const float* __restrict__ x,  const float* __restrict__ W1,
        const float* __restrict__ b1, const float* __restrict__ gamma,
        const float* __restrict__ beta, const float* __restrict__ Ws,
        const float* __restrict__ bs,
        __bf16* __restrict__ hbuf, float* __restrict__ skipbuf) {
    __shared__ float xs[DIM];
    __shared__ float hs[DIM];
    __shared__ float red[DIM];
    const int row = blockIdx.x, c = threadIdx.x;
    xs[c] = x[(size_t)row * DIM + c];
    __syncthreads();
    float acc = b1[c];
#pragma unroll 8
    for (int k = 0; k < DIM; ++k) acc = fmaf(xs[k], W1[k * DIM + c], acc);
    float e = acc > 0.f ? acc : (__expf(acc) - 1.f);     // ELU
    // mean
    red[c] = e; __syncthreads();
#pragma unroll
    for (int s = 64; s > 0; s >>= 1) { if (c < s) red[c] += red[c + s]; __syncthreads(); }
    float mu = red[0] * (1.f / DIM);
    __syncthreads();
    float d = e - mu;
    red[c] = d * d; __syncthreads();
#pragma unroll
    for (int s = 64; s > 0; s >>= 1) { if (c < s) red[c] += red[c + s]; __syncthreads(); }
    float var = red[0] * (1.f / DIM);
    float h = d * rsqrtf(var + 1e-5f) * gamma[c] + beta[c];
    hbuf[(size_t)row * DIM + c] = (__bf16)h;
    __syncthreads();
    hs[c] = h;
    __syncthreads();
    float sk = bs[c];
#pragma unroll 8
    for (int k = 0; k < DIM; ++k) sk = fmaf(hs[k], Ws[k * DIM + c], sk);
    skipbuf[(size_t)row * DIM + c] = sk;
}

// ---------- K2: Y = h @ [Wq|Wk|Wv] + bias   (WMMA bf16, f32 accum) ----------
__global__ __launch_bounds__(128) void qkv_gemm(
        const __bf16* __restrict__ h, const __bf16* __restrict__ WT,
        const float* __restrict__ bias, __bf16* __restrict__ Y) {
    const int lane = threadIdx.x & 31;
    const int wave = threadIdx.x >> 5;
    const int lr = lane & 15, hi = lane >> 4;
    const int r0 = blockIdx.x * 16;
    const int c0 = blockIdx.y * 64 + wave * 16;

    // A operand: rows of h (16x32 per K-step), contiguous K per lane half
    v16bf a[4];
    const __bf16* hrow = h + (size_t)(r0 + lr) * DIM;
#pragma unroll
    for (int i = 0; i < 4; ++i) {
        a[i] = *(const v16bf*)(hrow + i * 32 + hi * 16);
    }
    // B operand: column c of W, contiguous K (col-major WT)
    const __bf16* wcol = WT + (size_t)(c0 + lr) * DIM;
    v8f acc = vzero8();
#pragma unroll
    for (int i = 0; i < 4; ++i) {
        v16bf b = *(const v16bf*)(wcol + i * 32 + hi * 16);
        acc = __builtin_amdgcn_wmma_f32_16x16x32_bf16(false, a[i], false, b,
                                                      (short)0, acc, false, false);
    }
    float bv = bias[c0 + lr];
#pragma unroll
    for (int j = 0; j < 8; ++j) {
        int rr = r0 + j + 8 * hi;
        Y[(size_t)rr * QKVN + c0 + lr] = (__bf16)(acc[j] + bv);
    }
}

// ---------- K3: vT[h*128+d][n] = Y[n][1024 + h*128 + d] ----------
__global__ void transpose_v(const __bf16* __restrict__ Y, __bf16* __restrict__ vT) {
    __shared__ __bf16 tile[32][33];
    const int n0 = blockIdx.x * 32;     // node
    const int c0 = blockIdx.y * 32;     // v-channel (0..511)
    const int tx = threadIdx.x, ty = threadIdx.y;   // 32 x 8
#pragma unroll
    for (int t = 0; t < 4; ++t) {
        int r = ty + t * 8;
        tile[r][tx] = Y[(size_t)(n0 + r) * QKVN + 1024 + c0 + tx];
    }
    __syncthreads();
#pragma unroll
    for (int t = 0; t < 4; ++t) {
        int r = ty + t * 8;
        vT[(size_t)(c0 + r) * NNODES + n0 + tx] = tile[tx][r];
    }
}

// ---------- K4: flash attention over complete graph, 4 heads per block ----------
__global__ __launch_bounds__(128) void attn_kernel(
        const __bf16* __restrict__ Y, const __bf16* __restrict__ vT,
        const float* __restrict__ skip, float* __restrict__ out) {
    __shared__ float osum[HEADS][16 * DIM];          // per-head staging (deterministic)
    const int tid  = threadIdx.x;
    const int lane = tid & 31, head = tid >> 5;
    const int lr = lane & 15, hi = lane >> 4;
    const int q0 = blockIdx.x * 16;

    // Q^T B-operands (kept resident): column = query, K = channel (contiguous)
    v16bf bq[4];
    const __bf16* qrow = Y + (size_t)(q0 + lr) * QKVN + head * DIM;
#pragma unroll
    for (int i = 0; i < 4; ++i) bq[i] = *(const v16bf*)(qrow + i * 32 + hi * 16);

    v8f o[8];
#pragma unroll
    for (int t = 0; t < 8; ++t) o[t] = vzero8();
    float m = -3.0e38f, l = 0.f;

    const __bf16* kbase = Y + 512 + head * DIM;
    const __bf16* vbase = vT + (size_t)head * DIM * NNODES;

    for (int s0 = 0; s0 < NNODES; s0 += 32) {
        // S^T = K_tile @ Q^T : lane = query, VGPR = source
        v8f s[2]; s[0] = vzero8(); s[1] = vzero8();
#pragma unroll
        for (int t = 0; t < 2; ++t) {
            const __bf16* krow = kbase + (size_t)(s0 + t * 16 + lr) * QKVN;
#pragma unroll
            for (int i = 0; i < 4; ++i) {
                v16bf ak = *(const v16bf*)(krow + i * 32 + hi * 16);
                s[t] = __builtin_amdgcn_wmma_f32_16x16x32_bf16(false, ak, false, bq[i],
                                                               (short)0, s[t], false, false);
            }
        }
        // online softmax: each lane owns 16 of the 32 scores of its query
        float bm = s[0][0];
#pragma unroll
        for (int j = 1; j < 8; ++j) bm = fmaxf(bm, s[0][j]);
#pragma unroll
        for (int j = 0; j < 8; ++j) bm = fmaxf(bm, s[1][j]);
        bm = fmaxf(bm, __shfl_xor(bm, 16, 32));
        float mn    = fmaxf(m, bm);
        float alpha = __expf(m - mn);

        // P goes directly into A-operand layout (no cross-lane movement needed)
        v16bf ap;
        float ps = 0.f;
#pragma unroll
        for (int j = 0; j < 8; ++j) { float p = __expf(s[0][j] - mn); ps += p; ap[j]     = (__bf16)p; }
#pragma unroll
        for (int j = 0; j < 8; ++j) { float p = __expf(s[1][j] - mn); ps += p; ap[8 + j] = (__bf16)p; }
        ps += __shfl_xor(ps, 16, 32);
        l = l * alpha + ps;
        m = mn;

        // rescale output accumulators (query index lives in VGPR slot there)
        if (__ballot(alpha < 1.f) != 0ull) {
            float av[8];
#pragma unroll
            for (int j = 0; j < 8; ++j) av[j] = __shfl(alpha, j + 8 * hi, 32);
#pragma unroll
            for (int t = 0; t < 8; ++t)
#pragma unroll
                for (int j = 0; j < 8; ++j) o[t][j] *= av[j];
        }
        // O += P @ V : B operand from vT (table-order sources per lane)
#pragma unroll
        for (int t = 0; t < 8; ++t) {
            const __bf16* vrow = vbase + (size_t)(t * 16 + lr) * NNODES + s0 + hi * 8;
            v8bf vlo = *(const v8bf*)vrow;
            v8bf vhb = *(const v8bf*)(vrow + 16);
            v16bf bv = __builtin_shufflevector(vlo, vhb, 0,1,2,3,4,5,6,7,8,9,10,11,12,13,14,15);
            o[t] = __builtin_amdgcn_wmma_f32_16x16x32_bf16(false, ap, false, bv,
                                                           (short)0, o[t], false, false);
        }
    }

    // finalize: divide by l, apply 1/HEADS, stage per head
    float linv = 0.25f / l;
    float lv[8];
#pragma unroll
    for (int j = 0; j < 8; ++j) lv[j] = __shfl(linv, j + 8 * hi, 32);
#pragma unroll
    for (int t = 0; t < 8; ++t)
#pragma unroll
        for (int j = 0; j < 8; ++j) {
            int q = j + 8 * hi;
            int d = t * 16 + lr;
            osum[head][q * DIM + d] = o[t][j] * lv[j];
        }
    __syncthreads();
    for (int i = tid; i < 16 * DIM; i += 128) {
        int q = i >> 7, d = i & 127;
        float r = osum[0][i] + osum[1][i] + osum[2][i] + osum[3][i];
        out[(size_t)(q0 + q) * DIM + d] = r + skip[(size_t)(q0 + q) * DIM + d];
    }
}

// ---------- launcher ----------
extern "C" void kernel_launch(void* const* d_in, const int* in_sizes, int n_in,
                              void* d_out, int out_size, void* d_ws, size_t ws_size,
                              hipStream_t stream) {
    const float* x     = (const float*)d_in[0];
    const float* W1    = (const float*)d_in[1];
    const float* b1    = (const float*)d_in[2];
    const float* gamma = (const float*)d_in[3];
    const float* beta  = (const float*)d_in[4];
    const float* Wq    = (const float*)d_in[5];
    const float* bq    = (const float*)d_in[6];
    const float* Wk    = (const float*)d_in[7];
    const float* bk    = (const float*)d_in[8];
    const float* Wv    = (const float*)d_in[9];
    const float* bv    = (const float*)d_in[10];
    const float* Ws    = (const float*)d_in[11];
    const float* bs    = (const float*)d_in[12];

    char* ws = (char*)d_ws;
    __bf16* WT      = (__bf16*)(ws + OFF_WT);
    float*  biasall = (float*) (ws + OFF_BIAS);
    __bf16* hbuf    = (__bf16*)(ws + OFF_H);
    float*  skipb   = (float*) (ws + OFF_SKIP);
    __bf16* Y       = (__bf16*)(ws + OFF_Y);
    __bf16* vT      = (__bf16*)(ws + OFF_VT);
    float*  out     = (float*)d_out;

    pack_weights<<<dim3((QKVN * DIM + 255) / 256), 256, 0, stream>>>(
        Wq, Wk, Wv, bq, bk, bv, WT, biasall);
    row_ln_skip<<<dim3(NNODES), 128, 0, stream>>>(
        x, W1, b1, gamma, beta, Ws, bs, hbuf, skipb);
    qkv_gemm<<<dim3(NNODES / 16, QKVN / 64), 128, 0, stream>>>(hbuf, WT, biasall, Y);
    transpose_v<<<dim3(NNODES / 32, 512 / 32), dim3(32, 8), 0, stream>>>(Y, vT);
    attn_kernel<<<dim3(NNODES / 16), 128, 0, stream>>>(Y, vT, skipb, out);
}
